// WindowAttention_88553635709738
// MI455X (gfx1250) — matrix operand order repaired and checked by reading, not compiled, not run on hardware
//
#include <hip/hip_runtime.h>

#define DIM   128
#define HEADS 4
#define NTOK  49
#define HD    32

typedef __attribute__((ext_vector_type(16))) _Float16 v16h;
typedef __attribute__((ext_vector_type(8)))  _Float16 v8h;
typedef __attribute__((ext_vector_type(8)))  float    v8f;

#define WMMA_F16(a, b, c) \
  __builtin_amdgcn_wmma_f32_16x16x32_f16(false, (a), false, (b), (short)0, (c), false, false)

// d_ws layout (bytes):
//   [0, 65536)        : padded bias  float[4][64][64]  (masked cols = -1e30)
//   [65536, 163840)   : qkv_w as f16 [384][128]
//   [163840, 196608)  : proj_w as f16 [128][128]
#define WS_BIAS_OFF   0
#define WS_QKVW_OFF   65536
#define WS_PROJW_OFF  163840

// A fragment (16x32 f16, M x K): lane (hl,lp) holds row lp;
// halves 0..15 -> K = k0 + hl*16 + i
static __device__ __forceinline__ v16h load_a_lds(const _Float16* S, int ld,
                                                  int m0, int k0, int lp, int hl) {
  const _Float16* p = S + (m0 + lp) * ld + k0 + hl * 16;
  v8h lo = *(const v8h*)(p);
  v8h hi = *(const v8h*)(p + 8);
  v16h a;
#pragma unroll
  for (int i = 0; i < 8; ++i) { a[i] = lo[i]; a[i + 8] = hi[i]; }
  return a;
}

// B fragment (32x16 f16, K x N) where B column n is row n of a row-major f16 array:
// lane (hl,lp) holds column n0+lp, K halves = k0 + hl*16 + 0..15 (one 32B load)
static __device__ __forceinline__ v16h load_b_h(const _Float16* __restrict__ S, int ld,
                                                int n0, int k0, int lp, int hl) {
  return *(const v16h*)(S + (size_t)(n0 + lp) * ld + k0 + hl * 16);
}

// ---------------- prelude kernels (run once, tiny) ----------------

__global__ __launch_bounds__(256) void prep_weights_kernel(
    const float* __restrict__ w, _Float16* __restrict__ wh, int n) {
  int i = blockIdx.x * 256 + threadIdx.x;
  if (i < n) wh[i] = (_Float16)w[i];
}

__global__ __launch_bounds__(256) void prep_bias_kernel(
    const float* __restrict__ bias_table, const int* __restrict__ rel_index,
    float* __restrict__ biasp) {
  int i = blockIdx.x * 256 + threadIdx.x;  // over HEADS*64*64
  if (i < HEADS * 64 * 64) {
    int h = i >> 12, q = (i >> 6) & 63, k = i & 63;
    float v;
    if (k >= NTOK)      v = -1e30f;  // masked key column -> exp() = 0
    else if (q >= NTOK) v = 0.f;     // padded query row (unused downstream)
    else                v = bias_table[rel_index[q * NTOK + k] * HEADS + h];
    biasp[i] = v;
  }
}

// ---------------- main fused window-attention kernel ----------------
// LDS budget: 80 KB -> 4 blocks (32 waves) per 320 KB WGP.

__global__ __launch_bounds__(256) void win_attn_kernel(
    const float* __restrict__ x,
    const _Float16* __restrict__ qkv_wh,   // [384][128] f16
    const float* __restrict__ qkv_b,
    const _Float16* __restrict__ proj_wh,  // [128][128] f16
    const float* __restrict__ proj_b,
    const float* __restrict__ biasp,       // [4][64][64] f32 padded/masked
    float* __restrict__ out) {
  extern __shared__ _Float16 smem[];
  _Float16* xs   = smem;          // 64 x 128  x(f16); aliased as os after stage 1
  _Float16* os   = smem;          // 64 x 128  attention output (pre-proj)
  _Float16* qs   = smem + 8192;   // 64 x 128  q pre-scaled
  _Float16* ks   = smem + 16384;  // 64 x 128  k
  _Float16* vT   = smem + 24576;  // 128 x 64  v transposed
  _Float16* atts = smem + 32768;  // 8 waves x (16 x 64) softmaxed attn (wave-private)

  const int tid  = threadIdx.x;
  const int wv   = tid >> 5;  // wave 0..7
  const int lane = tid & 31;
  const int lp   = lane & 15;
  const int hl   = lane >> 4;
  const int blk  = blockIdx.x;
  const float scale = 0.17677669529663687f;  // 1/sqrt(32)

  // ---- stage 0: x -> f16 LDS via float4 loads (pad rows 49..63 with zero) ----
  {
    const float4* xg = (const float4*)(x + (size_t)blk * NTOK * DIM);
    for (int i = tid; i < 64 * 32; i += 256) {  // 32 float4 per row
      int r = i >> 5, c4 = i & 31;
      float4 val = (r < NTOK) ? xg[r * 32 + c4] : make_float4(0.f, 0.f, 0.f, 0.f);
      _Float16* d = xs + r * 128 + c4 * 4;
      d[0] = (_Float16)val.x; d[1] = (_Float16)val.y;
      d[2] = (_Float16)val.z; d[3] = (_Float16)val.w;
    }
  }
  __syncthreads();

  // ---- stage 1: QKV GEMM, three statically-routed segments (no branches) ----
  // t = wv + 8*j  ->  nt = wv (wave-invariant: B frags stay in registers), mt = j
  // Q segment: cols 0..127 of qkv -> qs (pre-scaled)
#pragma unroll 1
  for (int j = 0; j < 4; ++j) {
    int t = wv + 8 * j, mt = t >> 3, nt = t & 7;
    v8f acc = {};
#pragma unroll
    for (int kk = 0; kk < 4; ++kk) {
      v16h a = load_a_lds(xs, 128, mt * 16, kk * 32, lp, hl);
      v16h b = load_b_h(qkv_wh, 128, nt * 16, kk * 32, lp, hl);
      acc = WMMA_F16(a, b, acc);
    }
    int cg = nt * 16 + lp;
    float bb = qkv_b[cg];
#pragma unroll
    for (int v = 0; v < 8; ++v)
      qs[(mt * 16 + v + 8 * hl) * 128 + cg] = (_Float16)((acc[v] + bb) * scale);
  }
  // K segment: cols 128..255 -> ks
#pragma unroll 1
  for (int j = 0; j < 4; ++j) {
    int t = wv + 8 * j, mt = t >> 3, nt = t & 7;
    v8f acc = {};
#pragma unroll
    for (int kk = 0; kk < 4; ++kk) {
      v16h a = load_a_lds(xs, 128, mt * 16, kk * 32, lp, hl);
      v16h b = load_b_h(qkv_wh + 128 * 128, 128, nt * 16, kk * 32, lp, hl);
      acc = WMMA_F16(a, b, acc);
    }
    int cg = nt * 16 + lp;
    float bb = qkv_b[128 + cg];
#pragma unroll
    for (int v = 0; v < 8; ++v)
      ks[(mt * 16 + v + 8 * hl) * 128 + cg] = (_Float16)(acc[v] + bb);
  }
  // V segment: cols 256..383 -> vT (transposed store)
#pragma unroll 1
  for (int j = 0; j < 4; ++j) {
    int t = wv + 8 * j, mt = t >> 3, nt = t & 7;
    v8f acc = {};
#pragma unroll
    for (int kk = 0; kk < 4; ++kk) {
      v16h a = load_a_lds(xs, 128, mt * 16, kk * 32, lp, hl);
      v16h b = load_b_h(qkv_wh + 256 * 128, 128, nt * 16, kk * 32, lp, hl);
      acc = WMMA_F16(a, b, acc);
    }
    int cg = nt * 16 + lp;
    float bb = qkv_b[256 + cg];
#pragma unroll
    for (int v = 0; v < 8; ++v)
      vT[cg * 64 + (mt * 16 + v + 8 * hl)] = (_Float16)(acc[v] + bb);
  }
  __syncthreads();  // xs dead from here on; its space becomes os

  // ---- stage 2: attention, one (head, q-tile) strip per iteration ----
  _Float16* strip = atts + wv * 16 * 64;  // wave-private, reused across s
#pragma unroll 1
  for (int s = wv; s < 16; s += 8) {
    int h = s >> 2, mt = s & 3;
    // q @ k^T : A = q rows (16 x 32), B column = k row (contiguous d)
    v16h aq = load_a_lds(qs, 128, mt * 16, h * HD, lp, hl);
    v8f att[4];
#pragma unroll
    for (int kt = 0; kt < 4; ++kt) {
      v16h bk = load_b_h(ks, 128, kt * 16, h * HD, lp, hl);
      v8f z = {};
      att[kt] = WMMA_F16(aq, bk, z);
    }
    // branchless bias + mask from padded table (L2-resident)
    const float* bp = biasp + ((h * 64) + mt * 16 + 8 * hl) * 64 + lp;
#pragma unroll
    for (int v = 0; v < 8; ++v) {
#pragma unroll
      for (int kt = 0; kt < 4; ++kt)
        att[kt][v] += bp[v * 64 + kt * 16];
    }
    // row softmax: each row lives across 16 lanes of one half (xor<16 stays in-half)
#pragma unroll
    for (int v = 0; v < 8; ++v) {
      float m = fmaxf(fmaxf(att[0][v], att[1][v]), fmaxf(att[2][v], att[3][v]));
#pragma unroll
      for (int o = 1; o < 16; o <<= 1) m = fmaxf(m, __shfl_xor(m, o, 32));
      float e0 = __expf(att[0][v] - m), e1 = __expf(att[1][v] - m);
      float e2 = __expf(att[2][v] - m), e3 = __expf(att[3][v] - m);
      float sum = e0 + e1 + e2 + e3;
#pragma unroll
      for (int o = 1; o < 16; o <<= 1) sum += __shfl_xor(sum, o, 32);
      float rs = 1.0f / sum;
      int row = v + 8 * hl;
      strip[row * 64 +  0 + lp] = (_Float16)(e0 * rs);
      strip[row * 64 + 16 + lp] = (_Float16)(e1 * rs);
      strip[row * 64 + 32 + lp] = (_Float16)(e2 * rs);
      strip[row * 64 + 48 + lp] = (_Float16)(e3 * rs);
    }
    // attn @ v : strip[16x64] @ v[64x32]; B column d = vT row (contiguous tokens)
    v16h a0 = load_a_lds(strip, 64, 0, 0, lp, hl);
    v16h a1 = load_a_lds(strip, 64, 0, 32, lp, hl);
#pragma unroll
    for (int ntl = 0; ntl < 2; ++ntl) {
      v8f o = {};
      v16h b0 = load_b_h(vT, 64, h * HD + ntl * 16, 0, lp, hl);
      o = WMMA_F16(a0, b0, o);
      v16h b1 = load_b_h(vT, 64, h * HD + ntl * 16, 32, lp, hl);
      o = WMMA_F16(a1, b1, o);
#pragma unroll
      for (int v = 0; v < 8; ++v)
        os[(mt * 16 + v + 8 * hl) * 128 + h * HD + ntl * 16 + lp] = (_Float16)o[v];
    }
  }
  __syncthreads();

  // ---- stage 3: projection GEMM  out[49x128] = os[64x128] @ proj_w^T + proj_b ----
#pragma unroll 1
  for (int j = 0; j < 4; ++j) {
    int t = wv + 8 * j, mt = t >> 3, nt = t & 7;
    v8f acc = {};
#pragma unroll
    for (int kk = 0; kk < 4; ++kk) {
      v16h a = load_a_lds(os, 128, mt * 16, kk * 32, lp, hl);
      v16h b = load_b_h(proj_wh, 128, nt * 16, kk * 32, lp, hl);
      acc = WMMA_F16(a, b, acc);
    }
    int cg = nt * 16 + lp;
    float pb = proj_b[cg];
#pragma unroll
    for (int v = 0; v < 8; ++v) {
      int r = mt * 16 + v + 8 * hl;
      if (r < NTOK)
        out[((size_t)blk * NTOK + r) * DIM + cg] = acc[v] + pb;
    }
  }
}

extern "C" void kernel_launch(void* const* d_in, const int* in_sizes, int n_in,
                              void* d_out, int out_size, void* d_ws, size_t ws_size,
                              hipStream_t stream) {
  const float* x      = (const float*)d_in[0];
  const float* qkv_w  = (const float*)d_in[1];
  const float* qkv_b  = (const float*)d_in[2];
  const float* proj_w = (const float*)d_in[3];
  const float* proj_b = (const float*)d_in[4];
  const float* btab   = (const float*)d_in[5];
  const int*   ridx   = (const int*)d_in[6];
  float* out = (float*)d_out;

  float*    biasp   = (float*)((char*)d_ws + WS_BIAS_OFF);
  _Float16* qkv_wh  = (_Float16*)((char*)d_ws + WS_QKVW_OFF);
  _Float16* proj_wh = (_Float16*)((char*)d_ws + WS_PROJW_OFF);

  int B_ = in_sizes[0] / (NTOK * DIM);

  hipLaunchKernelGGL(prep_weights_kernel, dim3((384 * 128 + 255) / 256), dim3(256),
                     0, stream, qkv_w, qkv_wh, 384 * 128);
  hipLaunchKernelGGL(prep_weights_kernel, dim3((128 * 128 + 255) / 256), dim3(256),
                     0, stream, proj_w, proj_wh, 128 * 128);
  hipLaunchKernelGGL(prep_bias_kernel, dim3((HEADS * 64 * 64 + 255) / 256), dim3(256),
                     0, stream, btab, ridx, biasp);

  // 8192 (xs/os) + 8192 (qs) + 8192 (ks) + 8192 (vT) + 8192 (atts) halves = 80 KB
  size_t shmem = (size_t)40960 * sizeof(_Float16);
  hipLaunchKernelGGL(win_attn_kernel, dim3(B_), dim3(256), shmem, stream,
                     x, qkv_wh, qkv_b, proj_wh, proj_b, biasp, out);
}
